// Generator_40346922779109
// MI455X (gfx1250) — matrix operand order repaired and checked by reading, not compiled, not run on hardware
//
#include <hip/hip_runtime.h>
#include <hip/hip_bf16.h>
#include <math.h>

typedef __attribute__((ext_vector_type(16))) _Float16 v16h_t;
typedef __attribute__((ext_vector_type(8)))  _Float16 v8h_t;
typedef __attribute__((ext_vector_type(8)))  float    v8f_t;
typedef __attribute__((ext_vector_type(4)))  float    v4f_t;

#define TPB 256

__device__ __forceinline__ float lrelu_f(float v, float s) { return v >= 0.f ? v : s * v; }

// ---------------------------------------------------------------------------
// LDS-staged WMMA f16 GEMM.  D[b] = act( A[b] (MxK) @ B[b] (KxN) + bias + nw*noise )
// A row-major (lda = ldA, K contiguous); B row-major (ldb = ldB, N contiguous).
// Block tile 64x128x32, 8 waves (4x2), each wave 16x64 via 4 WMMAs per K-step.
// transD: 0 -> D[row*ldD+col], 1 -> D[col*ldD+row]. M%64==0, N%128==0, K%32==0.
// ---------------------------------------------------------------------------
__global__ void gemm_wmma_t(const _Float16* __restrict__ A, int ldA, int bsA,
                            const _Float16* __restrict__ Bp, int ldB, int bsB,
                            _Float16* __restrict__ outH, float* __restrict__ outF,
                            int ldD, int bsD, int transD,
                            const float* __restrict__ bias,
                            const float* __restrict__ noise, const float* __restrict__ nwPtr,
                            int M, int N, int K, int act)
{
    __shared__ _Float16 As[64 * 32];    // [m][k]
    __shared__ _Float16 Bs[128 * 32];   // [n][k] (transposed on fill)
    const int t    = threadIdx.x;
    const int lane = t & 31;
    const int w    = t >> 5;
    const int wm   = w >> 1, wn = w & 1;
    const int hl   = lane >> 4, ml = lane & 15;
    const int nBlocksN = N >> 7;
    const int bM = blockIdx.x / nBlocksN, bN = blockIdx.x % nBlocksN;
    const int b  = blockIdx.y;
    const int m0 = bM << 6, n0 = bN << 7;

    const _Float16* Ab = A  + (size_t)b * bsA;
    const _Float16* Bb = Bp + (size_t)b * bsB;

    const int aRow = t >> 2, aCol = (t & 3) << 3;   // 64 rows x 4 16B chunks

    v8f_t acc[4] = {};
    for (int k0 = 0; k0 < K; k0 += 32) {
        // A tile: one 16B load per thread
        *(v4f_t*)(As + aRow * 32 + aCol) =
            *(const v4f_t*)(Ab + (size_t)(m0 + aRow) * ldA + k0 + aCol);
        // B tile: 2x 16B loads per thread, transposed into [n][k]
#pragma unroll
        for (int c = 0; c < 2; ++c) {
            int li = t + (c << 8);                  // 512 chunks of 8 halfs
            int k = li >> 4, nc = (li & 15) << 3;
            v4f_t vv = *(const v4f_t*)(Bb + (size_t)(k0 + k) * ldB + n0 + nc);
            const _Float16* hp = (const _Float16*)&vv;
#pragma unroll
            for (int j = 0; j < 8; ++j) Bs[(nc + j) * 32 + k] = hp[j];
        }
        __syncthreads();

        v16h_t af;
        {
            const int base = (wm * 16 + ml) * 32 + hl * 8;
            v8h_t lo = *(const v8h_t*)(As + base);
            v8h_t hi = *(const v8h_t*)(As + base + 16);
#pragma unroll
            for (int j = 0; j < 8; ++j) { af[j] = lo[j]; af[8 + j] = hi[j]; }
        }
#pragma unroll
        for (int sub = 0; sub < 4; ++sub) {
            const int base = (wn * 64 + sub * 16 + ml) * 32 + hl * 8;
            v8h_t lo = *(const v8h_t*)(Bs + base);
            v8h_t hi = *(const v8h_t*)(Bs + base + 16);
            v16h_t bf;
#pragma unroll
            for (int j = 0; j < 8; ++j) { bf[j] = lo[j]; bf[8 + j] = hi[j]; }
            acc[sub] = __builtin_amdgcn_wmma_f32_16x16x32_f16(
                false, af, false, bf, (short)0, acc[sub], false, false);
        }
        __syncthreads();
    }

    const float nw = nwPtr ? nwPtr[0] : 0.f;
#pragma unroll
    for (int sub = 0; sub < 4; ++sub) {
        const int col = n0 + wn * 64 + sub * 16 + ml;
        const float nz = noise ? nw * noise[(size_t)b * N + col] : 0.f;
#pragma unroll
        for (int r = 0; r < 8; ++r) {
            const int row = m0 + wm * 16 + hl * 8 + r;
            float v = acc[sub][r];
            if (bias) v += bias[row];
            v += nz;
            if (act) v = lrelu_f(v, 0.2f);
            const size_t idx = (size_t)b * bsD +
                (transD ? (size_t)col * ldD + row : (size_t)row * ldD + col);
            if (outH) outH[idx] = (_Float16)v;
            if (outF) outF[idx] = v;
        }
    }
}

// ---------------------------------------------------------------------------
// 3x3 SAME conv = 9 shifted GEMMs with the same LDS tiling.
// X: [B,IC,H,W] f16.  Wh: pre-transposed weights [tap][OC][ICp] f16 (zero-padded).
// Y: [B,OC,H,W] f16 = lrelu(conv + bias, 0.2). OC%64==0, (H*W)%128==0, ICp%32==0.
// ---------------------------------------------------------------------------
__global__ void conv3x3_wmma(const _Float16* __restrict__ X, const _Float16* __restrict__ Wh,
                             const float* __restrict__ bias, _Float16* __restrict__ Y,
                             int IC, int ICp, int OC, int Hd, int Wd)
{
    __shared__ _Float16 As[64 * 32];
    __shared__ _Float16 Bs[128 * 32];
    const int t    = threadIdx.x;
    const int lane = t & 31;
    const int w    = t >> 5;
    const int wm   = w >> 1, wn = w & 1;
    const int hl   = lane >> 4, ml = lane & 15;
    const int Npix = Hd * Wd;
    const int nBlocksN = Npix >> 7;
    const int bM = blockIdx.x / nBlocksN, bN = blockIdx.x % nBlocksN;
    const int b  = blockIdx.y;
    const int m0 = bM << 6, n0 = bN << 7;
    const _Float16* Xb = X + (size_t)b * IC * Npix;
    const int aRow = t >> 2, aCol = (t & 3) << 3;

    v8f_t acc[4] = {};
    for (int tap = 0; tap < 9; ++tap) {
        const int dy = tap / 3 - 1, dx = tap % 3 - 1;
        const _Float16* Wt = Wh + (size_t)tap * OC * ICp;
        for (int k0 = 0; k0 < ICp; k0 += 32) {
            *(v4f_t*)(As + aRow * 32 + aCol) =
                *(const v4f_t*)(Wt + (size_t)(m0 + aRow) * ICp + k0 + aCol);
#pragma unroll
            for (int e = 0; e < 16; ++e) {
                int li = t + (e << 8);           // 4096 elements: [k][n] order
                int nn = li & 127, k = li >> 7;
                int p = n0 + nn;
                int y = p / Wd, x = p % Wd;
                int sy = y + dy, sx = x + dx;
                int ic = k0 + k;
                bool ok = (sy >= 0) && (sy < Hd) && (sx >= 0) && (sx < Wd) && (ic < IC);
                Bs[nn * 32 + k] = ok ? Xb[(size_t)ic * Npix + sy * Wd + sx] : (_Float16)0.f;
            }
            __syncthreads();

            v16h_t af;
            {
                const int base = (wm * 16 + ml) * 32 + hl * 8;
                v8h_t lo = *(const v8h_t*)(As + base);
                v8h_t hi = *(const v8h_t*)(As + base + 16);
#pragma unroll
                for (int j = 0; j < 8; ++j) { af[j] = lo[j]; af[8 + j] = hi[j]; }
            }
#pragma unroll
            for (int sub = 0; sub < 4; ++sub) {
                const int base = (wn * 64 + sub * 16 + ml) * 32 + hl * 8;
                v8h_t lo = *(const v8h_t*)(Bs + base);
                v8h_t hi = *(const v8h_t*)(Bs + base + 16);
                v16h_t bf;
#pragma unroll
                for (int j = 0; j < 8; ++j) { bf[j] = lo[j]; bf[8 + j] = hi[j]; }
                acc[sub] = __builtin_amdgcn_wmma_f32_16x16x32_f16(
                    false, af, false, bf, (short)0, acc[sub], false, false);
            }
            __syncthreads();
        }
    }
#pragma unroll
    for (int sub = 0; sub < 4; ++sub) {
        const int col = n0 + wn * 64 + sub * 16 + ml;
#pragma unroll
        for (int r = 0; r < 8; ++r) {
            const int row = m0 + wm * 16 + hl * 8 + r;
            float v = acc[sub][r] + bias[row];
            Y[(size_t)b * OC * Npix + (size_t)row * Npix + col] = (_Float16)lrelu_f(v, 0.2f);
        }
    }
}

// ----------------------------- small kernels -------------------------------
__global__ void conv_prep_k(const float* __restrict__ Wt, _Float16* __restrict__ Wh,
                            int IC, int ICp, int OC, int total)
{
    int i = blockIdx.x * blockDim.x + threadIdx.x;
    if (i >= total) return;                        // total = 9*OC*ICp
    int ic = i % ICp, oc = (i / ICp) % OC, tap = i / (ICp * OC);
    Wh[i] = (ic < IC) ? (_Float16)Wt[((size_t)oc * IC + ic) * 9 + tap] : (_Float16)0.f;
}

__global__ void pixelnorm_k(const float* __restrict__ z, float* __restrict__ out)
{
    int b = blockIdx.x;
    __shared__ float red[TPB];
    float s = 0.f;
    for (int i = threadIdx.x; i < 512; i += TPB) { float v = z[b * 512 + i]; s += v * v; }
    red[threadIdx.x] = s; __syncthreads();
    for (int st = TPB / 2; st > 0; st >>= 1) { if (threadIdx.x < st) red[threadIdx.x] += red[threadIdx.x + st]; __syncthreads(); }
    float r = rsqrtf(red[0] / 512.f + 1e-8f);
    for (int i = threadIdx.x; i < 512; i += TPB) out[b * 512 + i] = z[b * 512 + i] * r;
}

__global__ void linear_lrelu_k(const float* __restrict__ in, const float* __restrict__ W,
                               const float* __restrict__ bias, float* __restrict__ out, int n)
{
    int i = blockIdx.x * blockDim.x + threadIdx.x;
    if (i >= n) return;
    int b = i >> 9, o = i & 511;
    const float* ib = in + (b << 9);
    const float* wr = W + (size_t)o * 512;
    float acc = bias[o];
    for (int j = 0; j < 512; ++j) acc += ib[j] * wr[j];
    out[i] = acc >= 0.f ? acc : 0.01f * acc;
}

__global__ void mod_s_k(const float* __restrict__ wsty, const float* __restrict__ mw,
                        const float* __restrict__ mb, float* __restrict__ s, int IC, int total)
{
    int i = blockIdx.x * blockDim.x + threadIdx.x;
    if (i >= total) return;
    int b = i / IC, c = i % IC;
    const float* wb = wsty + b * 512;
    const float* mr = mw + (size_t)c * 512;
    float acc = 0.f;
    for (int j = 0; j < 512; ++j) acc += wb[j] * mr[j];
    s[i] = acc * 0.04419417382415922f + mb[c];   // 1/sqrt(512)
}

__global__ void build_wmod_k(const float* __restrict__ Wm, const float* __restrict__ s,
                             _Float16* __restrict__ out, int IC, int OC, int demod)
{
    int oc = blockIdx.x, b = blockIdx.y;
    float scale = rsqrtf((float)IC);
    __shared__ float red[TPB];
    float p = 0.f;
    for (int ic = threadIdx.x; ic < IC; ic += TPB) {
        float w = scale * Wm[(size_t)oc * IC + ic] * s[b * IC + ic];
        p += w * w;
    }
    red[threadIdx.x] = p; __syncthreads();
    for (int st = TPB / 2; st > 0; st >>= 1) { if (threadIdx.x < st) red[threadIdx.x] += red[threadIdx.x + st]; __syncthreads(); }
    float d = demod ? rsqrtf(red[0] + 1e-8f) : 1.f;
    for (int ic = threadIdx.x; ic < IC; ic += TPB)
        out[((size_t)b * OC + oc) * IC + ic] =
            (_Float16)(scale * Wm[(size_t)oc * IC + ic] * s[b * IC + ic] * d);
}

__global__ void lff_concat_k(const float* __restrict__ coords, const float* __restrict__ lw,
                             const float* __restrict__ lb, const float* __restrict__ cst,
                             _Float16* __restrict__ out, int total)
{
    int i = blockIdx.x * blockDim.x + threadIdx.x;
    if (i >= total) return;
    const int N = 16384;
    int n = i % N, c = (i / N) % 1024, b = i / (N * 1024);
    float v;
    if (c < 512) {
        v = sinf(lw[c * 3 + 0] * coords[(b * 3 + 0) * N + n] +
                 lw[c * 3 + 1] * coords[(b * 3 + 1) * N + n] +
                 lw[c * 3 + 2] * coords[(b * 3 + 2) * N + n] + lb[c]);
    } else {
        v = cst[(size_t)(c - 512) * N + n];
    }
    out[i] = (_Float16)v;
}

__global__ void pool_concat_k(const float* __restrict__ lat, const float* __restrict__ in2,
                              _Float16* __restrict__ out, int total)
{
    int i = blockIdx.x * blockDim.x + threadIdx.x;
    if (i >= total) return;  // [B,2,64,64]
    int x = i & 63, y = (i >> 6) & 63, c = (i >> 12) & 1, b = i >> 13;
    const float* src = (c == 0 ? lat : in2) + (size_t)b * 16384;
    int y0 = 2 * y, x0 = 2 * x;
    float m = fmaxf(fmaxf(src[y0 * 128 + x0], src[y0 * 128 + x0 + 1]),
                    fmaxf(src[(y0 + 1) * 128 + x0], src[(y0 + 1) * 128 + x0 + 1]));
    out[i] = (_Float16)m;
}

__global__ void maxpool_h_k(const _Float16* __restrict__ in, _Float16* __restrict__ out,
                            int C, int hIn, int total)
{
    int i = blockIdx.x * blockDim.x + threadIdx.x;
    if (i >= total) return;  // out [B,C,hIn/2,hIn/2]
    int ho = hIn >> 1;
    int x = i % ho, y = (i / ho) % ho, c = (i / (ho * ho)) % C, b = i / (ho * ho * C);
    const _Float16* src = in + ((size_t)b * C + c) * hIn * hIn;
    int y0 = 2 * y, x0 = 2 * x;
    float m = fmaxf(fmaxf((float)src[y0 * hIn + x0], (float)src[y0 * hIn + x0 + 1]),
                    fmaxf((float)src[(y0 + 1) * hIn + x0], (float)src[(y0 + 1) * hIn + x0 + 1]));
    out[i] = (_Float16)m;
}

__global__ void upsample2x_k(const _Float16* __restrict__ in, _Float16* __restrict__ out,
                             int C, int nIn, int total)
{
    int i = blockIdx.x * blockDim.x + threadIdx.x;
    if (i >= total) return;  // out [B,C,2n,2n], align_corners=True
    int on = 2 * nIn;
    int x = i % on, y = (i / on) % on, c = (i / (on * on)) % C, b = i / (on * on * C);
    float scale = (float)(nIn - 1) / (float)(on - 1);
    float py = y * scale, px = x * scale;
    int y0 = (int)floorf(py), x0 = (int)floorf(px);
    int y1 = min(y0 + 1, nIn - 1), x1 = min(x0 + 1, nIn - 1);
    float ty = py - y0, tx = px - x0;
    const _Float16* src = in + ((size_t)b * C + c) * nIn * nIn;
    float a00 = (float)src[y0 * nIn + x0], a01 = (float)src[y0 * nIn + x1];
    float a10 = (float)src[y1 * nIn + x0], a11 = (float)src[y1 * nIn + x1];
    float v = (a00 * (1.f - tx) + a01 * tx) * (1.f - ty) + (a10 * (1.f - tx) + a11 * tx) * ty;
    out[i] = (_Float16)v;
}

__global__ void softmax_rows_k(const float* __restrict__ S, _Float16* __restrict__ P, int cols)
{
    int row = blockIdx.x;
    const float* src = S + (size_t)row * cols;
    __shared__ float red[TPB];
    float mx = -3.0e38f;
    for (int j = threadIdx.x; j < cols; j += TPB) mx = fmaxf(mx, src[j]);
    red[threadIdx.x] = mx; __syncthreads();
    for (int st = TPB / 2; st > 0; st >>= 1) { if (threadIdx.x < st) red[threadIdx.x] = fmaxf(red[threadIdx.x], red[threadIdx.x + st]); __syncthreads(); }
    mx = red[0]; __syncthreads();
    float s = 0.f;
    for (int j = threadIdx.x; j < cols; j += TPB) s += expf(src[j] - mx);
    red[threadIdx.x] = s; __syncthreads();
    for (int st = TPB / 2; st > 0; st >>= 1) { if (threadIdx.x < st) red[threadIdx.x] += red[threadIdx.x + st]; __syncthreads(); }
    float inv = 1.f / red[0];
    for (int j = threadIdx.x; j < cols; j += TPB)
        P[(size_t)row * cols + j] = (_Float16)(expf(src[j] - mx) * inv);
}

// attnO is transposed: [B][pix][512]; xin/out are [B][512][1024]
__global__ void combine_attn_k(const float* __restrict__ ao, const _Float16* __restrict__ xin,
                               const float* __restrict__ gamma, _Float16* __restrict__ out, int total)
{
    int i = blockIdx.x * blockDim.x + threadIdx.x;
    if (i >= total) return;
    int pix = i & 1023, c = (i >> 10) & 511, b = i >> 19;
    float v = gamma[0] * ao[(size_t)b * 512 * 1024 + (size_t)pix * 512 + c] + (float)xin[i];
    out[i] = (_Float16)v;
}

__global__ void add_h_k(const _Float16* __restrict__ a, const _Float16* __restrict__ b,
                        _Float16* __restrict__ c, int total)
{
    int i = blockIdx.x * blockDim.x + threadIdx.x;
    if (i >= total) return;
    c[i] = (_Float16)((float)a[i] + (float)b[i]);
}

__global__ void cvt_h_k(const float* __restrict__ in, _Float16* __restrict__ out, int n)
{
    int i = blockIdx.x * blockDim.x + threadIdx.x;
    if (i < n) out[i] = (_Float16)in[i];
}

__global__ void zero_f_k(float* __restrict__ p, int n)
{
    int i = blockIdx.x * blockDim.x + threadIdx.x;
    if (i < n) p[i] = 0.f;
}

__global__ void to_rgb_gemv_k(const _Float16* __restrict__ x, const _Float16* __restrict__ w,
                              const float* __restrict__ biasPtr, const float* __restrict__ rin,
                              float* __restrict__ rout, int IC, int N, int total)
{
    int i = blockIdx.x * blockDim.x + threadIdx.x;
    if (i >= total) return;
    int b = i / N, n = i % N;
    float acc = rin[i] + biasPtr[0];
    const _Float16* xb = x + (size_t)b * IC * N + n;
    const _Float16* wb = w + (size_t)b * IC;
    for (int ic = 0; ic < IC; ++ic) acc += (float)wb[ic] * (float)xb[(size_t)ic * N];
    rout[i] = acc;
}

// ---------------------------------------------------------------------------
extern "C" void kernel_launch(void* const* d_in, const int* in_sizes, int n_in,
                              void* d_out, int out_size, void* d_ws, size_t ws_size,
                              hipStream_t stream)
{
    (void)in_sizes; (void)n_in; (void)out_size; (void)ws_size;
    constexpr int Bn = 2;
    constexpr int Np = 128 * 128;  // 16384

    const float* coords = (const float*)d_in[0];
    const float* latent = (const float*)d_in[1];
    const float* input2 = (const float*)d_in[2];
    const float* zlat   = (const float*)d_in[3];
    const float* inj    = (const float*)d_in[4];
    const float* lffw   = (const float*)d_in[5];
    const float* lffb   = (const float*)d_in[6];
    const float* cst    = (const float*)d_in[7];

    // ---- workspace carve (bump allocator, 256B aligned) ----
    char* wp = (char*)d_ws;
    auto alloc = [&](size_t bytes) -> void* {
        void* p = (void*)wp;
        wp += (bytes + 255) & ~(size_t)255;
        return p;
    };
    float*    wstyA = (float*)alloc((size_t)Bn * 512 * 4);
    float*    wstyB = (float*)alloc((size_t)Bn * 512 * 4);
    float*    sbuf  = (float*)alloc((size_t)Bn * 1024 * 4);
    _Float16* wmodH = (_Float16*)alloc((size_t)Bn * 512 * 1024 * 2);
    _Float16* wrgbH = (_Float16*)alloc((size_t)Bn * 512 * 2);
    _Float16* WqH   = (_Float16*)alloc((size_t)64 * 512 * 2);
    _Float16* WkH   = (_Float16*)alloc((size_t)64 * 512 * 2);
    _Float16* WvH   = (_Float16*)alloc((size_t)512 * 512 * 2);
    _Float16* WcH   = (_Float16*)alloc((size_t)9 * 512 * 512 * 2);   // conv weights (reused)
    _Float16* x0H   = (_Float16*)alloc((size_t)Bn * 1024 * Np * 2);
    _Float16* xA    = (_Float16*)alloc((size_t)Bn * 512 * Np * 2);
    _Float16* xB    = (_Float16*)alloc((size_t)Bn * 512 * Np * 2);
    _Float16* latA  = (_Float16*)alloc((size_t)Bn * 512 * Np * 2);
    _Float16* latB  = (_Float16*)alloc((size_t)Bn * 512 * Np * 2);
    _Float16* lat2  = (_Float16*)alloc((size_t)Bn * 2 * 64 * 64 * 2);
    _Float16* t64a  = (_Float16*)alloc((size_t)Bn * 512 * 4096 * 2);
    _Float16* t64b  = (_Float16*)alloc((size_t)Bn * 512 * 4096 * 2);
    _Float16* t32a  = (_Float16*)alloc((size_t)Bn * 512 * 1024 * 2);
    _Float16* t32b  = (_Float16*)alloc((size_t)Bn * 512 * 1024 * 2);
    _Float16* qH    = (_Float16*)alloc((size_t)Bn * 1024 * 64 * 2);   // Q^T [pix][64]
    _Float16* kH    = (_Float16*)alloc((size_t)Bn * 64 * 1024 * 2);   // K   [64][pix]
    _Float16* vH    = (_Float16*)alloc((size_t)Bn * 1024 * 512 * 2);  // V^T [pix][512]
    float*    scor  = (float*)alloc((size_t)Bn * 1024 * 1024 * 4);
    _Float16* attnH = (_Float16*)alloc((size_t)Bn * 1024 * 1024 * 2);
    float*    attnO = (float*)alloc((size_t)Bn * 1024 * 512 * 4);     // transposed [pix][512]
    float*    rgb32 = (float*)alloc((size_t)Bn * Np * 4);

    auto blocks = [](int total) { return (unsigned)((total + TPB - 1) / TPB); };

    auto gemm = [&](const _Float16* A, int ldA, int bsA,
                    const _Float16* Bp, int ldB, int bsB,
                    _Float16* oH, float* oF, int ldD, int bsD, int transD,
                    const float* bias, const float* noise, const float* nw,
                    int M, int Nn, int K, int act) {
        dim3 g((unsigned)((M / 64) * (Nn / 128)), (unsigned)Bn);
        gemm_wmma_t<<<g, TPB, 0, stream>>>(A, ldA, bsA, Bp, ldB, bsB, oH, oF, ldD, bsD,
                                           transD, bias, noise, nw, M, Nn, K, act);
    };
    auto conv = [&](const _Float16* X, const float* Wt, const float* bias,
                    _Float16* Y, int IC, int OC, int Hd, int Wd) {
        int ICp = (IC + 31) & ~31;
        conv_prep_k<<<blocks(9 * OC * ICp), TPB, 0, stream>>>(Wt, WcH, IC, ICp, OC, 9 * OC * ICp);
        dim3 g((unsigned)((OC / 64) * ((Hd * Wd) / 128)), (unsigned)Bn);
        conv3x3_wmma<<<g, TPB, 0, stream>>>(X, WcH, bias, Y, IC, ICp, OC, Hd, Wd);
    };

    // ---- style MLP: pixelnorm + 8 x (linear + lrelu 0.01) ----
    pixelnorm_k<<<Bn, TPB, 0, stream>>>(zlat, wstyA);
    float* scur = wstyA; float* snxt = wstyB;
    for (int i = 0; i < 8; ++i) {
        linear_lrelu_k<<<blocks(Bn * 512), TPB, 0, stream>>>(
            scur, (const float*)d_in[8 + 2 * i], (const float*)d_in[9 + 2 * i], snxt, Bn * 512);
        float* t = scur; scur = snxt; snxt = t;
    }
    const float* wsty = scur;

    // ---- LFF + const concat -> x0 [B,1024,16384] f16 ----
    lff_concat_k<<<blocks(Bn * 1024 * Np), TPB, 0, stream>>>(coords, lffw, lffb, cst, x0H, Bn * 1024 * Np);

    // ---- conv1: modulated 1x1 (1024 -> 512) + noise + lrelu ----
    mod_s_k<<<blocks(Bn * 1024), TPB, 0, stream>>>(wsty, (const float*)d_in[25], (const float*)d_in[26], sbuf, 1024, Bn * 1024);
    build_wmod_k<<<dim3(512, Bn), TPB, 0, stream>>>((const float*)d_in[24], sbuf, wmodH, 1024, 512, 1);
    gemm(wmodH, 1024, 512 * 1024, x0H, Np, 1024 * Np, xA, nullptr, Np, 512 * Np, 0,
         nullptr, inj, (const float*)d_in[27], 512, Np, 1024, 1);

    // ---- conditioning branch ----
    pool_concat_k<<<blocks(Bn * 2 * 64 * 64), TPB, 0, stream>>>(latent, input2, lat2, Bn * 2 * 64 * 64);
    conv(lat2, (const float*)d_in[28], (const float*)d_in[29], t64a, 2, 512, 64, 64);
    maxpool_h_k<<<blocks(Bn * 512 * 1024), TPB, 0, stream>>>(t64a, t32a, 512, 64, Bn * 512 * 1024);
    conv(t32a, (const float*)d_in[30], (const float*)d_in[31], t32b, 512, 512, 32, 32);

    // self-attention at 32x32 (N=1024, C=512, c8=64)
    cvt_h_k<<<blocks(64 * 512), TPB, 0, stream>>>((const float*)d_in[32], WqH, 64 * 512);
    cvt_h_k<<<blocks(64 * 512), TPB, 0, stream>>>((const float*)d_in[34], WkH, 64 * 512);
    cvt_h_k<<<blocks(512 * 512), TPB, 0, stream>>>((const float*)d_in[36], WvH, 512 * 512);
    // Q^T: store transposed [pix][64]
    gemm(WqH, 512, 0, t32b, 1024, 512 * 1024, qH, nullptr, 64, 64 * 1024, 1,
         (const float*)d_in[33], nullptr, nullptr, 64, 1024, 512, 0);
    // K: normal [64][pix]
    gemm(WkH, 512, 0, t32b, 1024, 512 * 1024, kH, nullptr, 1024, 64 * 1024, 0,
         (const float*)d_in[35], nullptr, nullptr, 64, 1024, 512, 0);
    // V^T: store transposed [pix][512]
    gemm(WvH, 512, 0, t32b, 1024, 512 * 1024, vH, nullptr, 512, 1024 * 512, 1,
         (const float*)d_in[37], nullptr, nullptr, 512, 1024, 512, 0);
    // scores[n][m] = sum_q Q^T[n][q] * K[q][m]
    gemm(qH, 64, 64 * 1024, kH, 1024, 64 * 1024, nullptr, scor, 1024, 1024 * 1024, 0,
         nullptr, nullptr, nullptr, 1024, 1024, 64, 0);
    softmax_rows_k<<<Bn * 1024, TPB, 0, stream>>>(scor, attnH, 1024);
    // outT[m][o] = sum_n attn[m][n] * V^T[n][o]
    gemm(attnH, 1024, 1024 * 1024, vH, 512, 1024 * 512, nullptr, attnO, 512, 1024 * 512, 0,
         nullptr, nullptr, nullptr, 1024, 512, 1024, 0);
    combine_attn_k<<<blocks(Bn * 512 * 1024), TPB, 0, stream>>>(attnO, t32b, (const float*)d_in[38], t32a, Bn * 512 * 1024);

    // up path: 32 -> 64 -> 128
    upsample2x_k<<<blocks(Bn * 512 * 4096), TPB, 0, stream>>>(t32a, t64a, 512, 32, Bn * 512 * 4096);
    conv(t64a, (const float*)d_in[39], (const float*)d_in[40], t64b, 512, 512, 64, 64);
    upsample2x_k<<<blocks(Bn * 512 * Np), TPB, 0, stream>>>(t64b, latA, 512, 64, Bn * 512 * Np);
    conv(latA, (const float*)d_in[41], (const float*)d_in[42], latB, 512, 512, 128, 128);

    // x = x + lat
    add_h_k<<<blocks(Bn * 512 * Np), TPB, 0, stream>>>(xA, latB, xB, Bn * 512 * Np);

    // ---- main synthesis loop ----
    zero_f_k<<<blocks(Bn * Np), TPB, 0, stream>>>(rgb32, Bn * Np);
    _Float16* curx = xB;
    _Float16* tmpx = xA;
    int ic = 512, nidx = 1;
    for (int i = 0; i < 6; ++i) {
        int oc = (i == 5) ? 256 : 512;
        int li = 43 + 4 * (2 * i);
        // styled conv A: ic -> oc
        mod_s_k<<<blocks(Bn * ic), TPB, 0, stream>>>(wsty, (const float*)d_in[li + 1], (const float*)d_in[li + 2], sbuf, ic, Bn * ic);
        build_wmod_k<<<dim3((unsigned)oc, Bn), TPB, 0, stream>>>((const float*)d_in[li], sbuf, wmodH, ic, oc, 1);
        gemm(wmodH, ic, oc * ic, curx, Np, ic * Np, tmpx, nullptr, Np, oc * Np, 0,
             nullptr, inj + (size_t)nidx * Bn * Np, (const float*)d_in[li + 3], oc, Np, ic, 1);
        ++nidx;
        // styled conv B: oc -> oc
        int lj = li + 4;
        mod_s_k<<<blocks(Bn * oc), TPB, 0, stream>>>(wsty, (const float*)d_in[lj + 1], (const float*)d_in[lj + 2], sbuf, oc, Bn * oc);
        build_wmod_k<<<dim3((unsigned)oc, Bn), TPB, 0, stream>>>((const float*)d_in[lj], sbuf, wmodH, oc, oc, 1);
        gemm(wmodH, oc, oc * oc, tmpx, Np, oc * Np, curx, nullptr, Np, oc * Np, 0,
             nullptr, inj + (size_t)nidx * Bn * Np, (const float*)d_in[lj + 3], oc, Np, oc, 1);
        ++nidx;
        // to_rgb (1 channel, no demod), accumulate skip
        int ri = 91 + 4 * i;
        mod_s_k<<<blocks(Bn * oc), TPB, 0, stream>>>(wsty, (const float*)d_in[ri + 1], (const float*)d_in[ri + 2], sbuf, oc, Bn * oc);
        build_wmod_k<<<dim3(1, Bn), TPB, 0, stream>>>((const float*)d_in[ri], sbuf, wrgbH, oc, 1, 0);
        float* rout = (i == 5) ? (float*)d_out : rgb32;
        to_rgb_gemv_k<<<blocks(Bn * Np), TPB, 0, stream>>>(curx, wrgbH, (const float*)d_in[ri + 3], rgb32, rout, oc, Np, Bn * Np);
        ic = oc;
    }
}
